// MultiHeadSelfAttention_39685497815603
// MI455X (gfx1250) — matrix operand order repaired and checked by reading, not compiled, not run on hardware
//
#include <hip/hip_runtime.h>
#include <hip/hip_bf16.h>

// ---------------------------------------------------------------------------
// MI455X (gfx1250) multi-head self-attention, bf16 WMMA pipeline:
//   0) f32->bf16 pre-convert of x, qkv_w, out_w (once; removes 48-64x
//      redundant per-tile conversions and enables async LDS staging)
//   1) qkv_gemm : xb[8192,1024] x qkv_wb^T + b -> Q,K,V bf16 [B,H,T,D]
//   2) attn     : flash-style attention per (b,h), 16 q-rows per wave
//   3) out_gemm : attn[8192,1024](bf16) x out_wb^T + b -> f32 output
// Matrix math: v_wmma_f32_16x16x32_bf16. Tile staging: GLOBAL_LOAD_ASYNC_TO_LDS
// (ASYNCcnt) for all bf16 operands.
// ---------------------------------------------------------------------------

#define EMBED 1024
#define HEADS 16
#define HDIM  64
#define SEQ   2048
#define NBATCH 4
#define HEAD_ELEMS (NBATCH * HEADS * SEQ * HDIM)   // 8388608 elements per Q/K/V

typedef __attribute__((ext_vector_type(16))) __bf16 bf16x16;
typedef __attribute__((ext_vector_type(8)))  float  f32x8;
typedef int v4i __attribute__((vector_size(16)));

union FragB { unsigned u[8]; bf16x16 v; };
union FragF { f32x8 v; float f[8]; };

__device__ __forceinline__ unsigned short cvt_bf16(float a) {
    unsigned ua = __float_as_uint(a);
    ua += 0x7FFFu + ((ua >> 16) & 1u);           // round-to-nearest-even
    return (unsigned short)(ua >> 16);
}
__device__ __forceinline__ unsigned cvt_bf16x2(float a, float b) {
    unsigned ua = __float_as_uint(a);
    unsigned ub = __float_as_uint(b);
    ua += 0x7FFFu + ((ua >> 16) & 1u);
    ub += 0x7FFFu + ((ub >> 16) & 1u);
    return (ua >> 16) | (ub & 0xFFFF0000u);
}

// Async global->LDS 16B copy (GLOBAL_LOAD_ASYNC_TO_LDS_B128, tracked by ASYNCcnt)
__device__ __forceinline__ void async_copy_b128(const void* gsrc, void* lds) {
    __builtin_amdgcn_global_load_async_to_lds_b128(
        (__attribute__((address_space(1))) v4i*)gsrc,
        (__attribute__((address_space(3))) v4i*)lds, 0, 0);
}
__device__ __forceinline__ void async_wait0() {
    __builtin_amdgcn_s_wait_asynccnt(0);
}

// ---------------------------------------------------------------------------
// Kernel 0: bulk f32 -> bf16 conversion (vectorized float4 -> 4x bf16)
// ---------------------------------------------------------------------------
__global__ __launch_bounds__(256)
void f32_to_bf16(const float* __restrict__ src, unsigned short* __restrict__ dst, int n4)
{
    const int i = blockIdx.x * 256 + threadIdx.x;
    if (i < n4) {
        const float4 f = ((const float4*)src)[i];
        uint2 o;
        o.x = cvt_bf16x2(f.x, f.y);
        o.y = cvt_bf16x2(f.z, f.w);
        ((uint2*)dst)[i] = o;
    }
}

// ---------------------------------------------------------------------------
// Kernel 1: QKV projection. Block tile 128(M) x 64(N), 8 waves, K-step 32.
// bf16 A/B tiles staged via async-to-LDS. Output scattered to [which][B,H,T,D].
// ---------------------------------------------------------------------------
__global__ __launch_bounds__(256)
void qkv_gemm(const unsigned short* __restrict__ xb,
              const unsigned short* __restrict__ wb,
              const float* __restrict__ bias, unsigned short* __restrict__ qkv)
{
    // row stride 20 uints = 80B: 16B-aligned chunks, 64-bank conflict-free
    __shared__ __align__(16) unsigned lA[128][20];
    __shared__ __align__(16) unsigned lB[64][20];

    const int tid  = threadIdx.x;
    const int lane = tid & 31, wave = tid >> 5;
    const int half = lane >> 4, col = lane & 15;
    const int m0 = blockIdx.y * 128;
    const int n0 = blockIdx.x * 64;

    FragF acc[4];
    const f32x8 fzero = {};
#pragma unroll
    for (int t = 0; t < 4; t++) acc[t].v = fzero;

    for (int k0 = 0; k0 < EMBED; k0 += 32) {
        // A tile 128x32 bf16 = 512 x 16B chunks (2 per thread)
#pragma unroll
        for (int i = 0; i < 2; i++) {
            int c = tid + i * 256;
            int row = c >> 2, q = c & 3;
            async_copy_b128(xb + (size_t)(m0 + row) * EMBED + k0 + q * 8, &lA[row][q * 4]);
        }
        // B tile 64x32 bf16 = 256 x 16B chunks (1 per thread)
        {
            int row = tid >> 2, q = tid & 3;
            async_copy_b128(wb + (size_t)(n0 + row) * EMBED + k0 + q * 8, &lB[row][q * 4]);
        }
        async_wait0();
        __syncthreads();

        FragB a;
        const int mr = wave * 16 + col;
#pragma unroll
        for (int j = 0; j < 8; j++) {           // A-layout pairs {8h .. 8h+7}
            a.u[j]     = lA[mr][j + 8 * half];
        }
#pragma unroll
        for (int t = 0; t < 4; t++) {
            FragB b; const int nr = t * 16 + col;
#pragma unroll
            for (int j = 0; j < 8; j++)          // B-layout pairs {8h .. 8h+7}
                b.u[j] = lB[nr][j + 8 * half];
            acc[t].v = __builtin_amdgcn_wmma_f32_16x16x32_bf16(
                false, a.v, false, b.v, (short)0, acc[t].v, false, false);
        }
        __syncthreads();
    }

    // epilogue: bias + scatter to [which][B,H,T,D] bf16
#pragma unroll
    for (int t = 0; t < 4; t++) {
        const int gn = n0 + t * 16 + col;
        const int which = gn >> 10, c = gn & 1023;
        const int head = c >> 6, d = c & 63;
        const float bv = bias[gn];
#pragma unroll
        for (int r = 0; r < 8; r++) {
            const int gm = m0 + wave * 16 + r + 8 * half;
            const int bb = gm >> 11, tq = gm & (SEQ - 1);
            size_t dst = (size_t)which * HEAD_ELEMS +
                         (((size_t)(bb * HEADS + head)) * SEQ + tq) * HDIM + d;
            qkv[dst] = cvt_bf16(acc[t].f[r] + bv);
        }
    }
}

// ---------------------------------------------------------------------------
// Kernel 2: flash attention. One block = 128 q-rows of one (b,h); each of the
// 8 waves owns 16 q-rows. 32-key K tile async-staged; V tile transposed in LDS.
// ---------------------------------------------------------------------------
__global__ __launch_bounds__(256)
void attn_kernel(const unsigned short* __restrict__ qkv,
                 const int* __restrict__ mask,
                 unsigned short* __restrict__ attn)
{
    __shared__ __align__(16) unsigned k_lds[32][36];   // [key][d-pair], 144B rows
    __shared__ unsigned short vt_lds[64][34];          // [d][key] (V transposed)
    __shared__ unsigned short p_lds[8][16][34];        // per-wave P tile 16x32

    const int tid = threadIdx.x, lane = tid & 31, wave = tid >> 5;
    const int half = lane >> 4, col = lane & 15;
    const int blk  = blockIdx.x;
    const int qblk = blk & 15;
    const int head = (blk >> 4) & 15;
    const int b    = blk >> 8;

    const size_t bh = (size_t)(b * HEADS + head) * SEQ * HDIM;
    const unsigned short* Ks = qkv + HEAD_ELEMS + bh;
    const unsigned* Qu = (const unsigned*)qkv + (bh >> 1);                   // row = 32 uints
    const unsigned* Vu = (const unsigned*)(qkv + 2 * HEAD_ELEMS) + (bh >> 1);

    const int q0 = qblk * 128 + wave * 16;

    // Q fragments (A layout): Qa = d[0..31], Qb = d[32..63]
    FragB qa, qb;
    {
        const unsigned* qrow = Qu + (size_t)(q0 + col) * 32;
#pragma unroll
        for (int j = 0; j < 8; j++) {
            qa.u[j]     = qrow[j + 8 * half];
            qb.u[j]     = qrow[16 + j + 8 * half];
        }
    }

    FragF accO[4];
    const f32x8 fzero = {};
#pragma unroll
    for (int j = 0; j < 4; j++) accO[j].v = fzero;
    float m_i[8], l_i[8];
#pragma unroll
    for (int r = 0; r < 8; r++) { m_i[r] = -1e30f; l_i[r] = 0.f; }

    const int maskBase = b * SEQ;
    const float NEG = -1e30f;

    for (int kt = 0; kt < SEQ; kt += 32) {
        // K tile [32 keys][64 bf16] = 256 x 16B async chunks (1 per thread)
        {
            int row = tid >> 3, q = tid & 7;
            async_copy_b128(Ks + (size_t)(kt + row) * HDIM + q * 8, &k_lds[row][q * 4]);
        }
        // V tile transposed: [d][key] (transpose needs per-element stores)
#pragma unroll
        for (int i = 0; i < 4; i++) {
            int idx = tid + i * 256;
            int key = idx >> 5, dp = idx & 31;
            unsigned vv = Vu[(size_t)(kt + key) * 32 + dp];
            vt_lds[dp * 2][key]     = (unsigned short)(vv & 0xFFFFu);
            vt_lds[dp * 2 + 1][key] = (unsigned short)(vv >> 16);
        }
        async_wait0();
        __syncthreads();

        // S = Q K^T for two 16-key subtiles (contraction d = 64 -> 2 WMMAs each)
        FragF s[2];
#pragma unroll
        for (int si = 0; si < 2; si++) {
            FragB ba, bb;
            const int key = si * 16 + col;
#pragma unroll
            for (int j = 0; j < 8; j++) {
                ba.u[j] = k_lds[key][8 * half + j];        // d[0..31]
                bb.u[j] = k_lds[key][16 + 8 * half + j];   // d[32..63]
            }
            s[si].v = __builtin_amdgcn_wmma_f32_16x16x32_bf16(
                false, qa.v, false, ba.v, (short)0, fzero, false, false);
            s[si].v = __builtin_amdgcn_wmma_f32_16x16x32_bf16(
                false, qb.v, false, bb.v, (short)0, s[si].v, false, false);
        }

        // mask + scale (each lane's column is a single key)
        const int mk0 = mask[maskBase + kt + col];
        const int mk1 = mask[maskBase + kt + 16 + col];
#pragma unroll
        for (int r = 0; r < 8; r++) {
            s[0].f[r] = mk0 ? s[0].f[r] * 0.125f : NEG;
            s[1].f[r] = mk1 ? s[1].f[r] * 0.125f : NEG;
        }

        // online softmax; row M = r + 8*half lives across 16 lanes of a half
        float p0[8], p1[8];
#pragma unroll
        for (int r = 0; r < 8; r++) {
            float mx = fmaxf(s[0].f[r], s[1].f[r]);
#pragma unroll
            for (int off = 8; off >= 1; off >>= 1)
                mx = fmaxf(mx, __shfl_xor(mx, off, 32));
            const float mn  = fmaxf(m_i[r], mx);
            const float fac = __expf(m_i[r] - mn);
            p0[r] = __expf(s[0].f[r] - mn);
            p1[r] = __expf(s[1].f[r] - mn);
            float rs = p0[r] + p1[r];
#pragma unroll
            for (int off = 8; off >= 1; off >>= 1)
                rs += __shfl_xor(rs, off, 32);
            l_i[r] = l_i[r] * fac + rs;
            m_i[r] = mn;
#pragma unroll
            for (int j = 0; j < 4; j++) accO[j].f[r] *= fac;
        }

        // C-layout P -> LDS -> A-layout fragment (per-wave region, no barrier)
#pragma unroll
        for (int r = 0; r < 8; r++) {
            p_lds[wave][r + 8 * half][col]      = cvt_bf16(p0[r]);
            p_lds[wave][r + 8 * half][16 + col] = cvt_bf16(p1[r]);
        }
        FragB pf;
        {
            const unsigned* prow = (const unsigned*)&p_lds[wave][col][0];
#pragma unroll
            for (int j = 0; j < 8; j++) {
                pf.u[j]     = prow[j + 8 * half];
            }
        }

        // O += P V over 4 d-tiles
#pragma unroll
        for (int j = 0; j < 4; j++) {
            FragB vf;
            const unsigned* vrow = (const unsigned*)&vt_lds[j * 16 + col][0];
#pragma unroll
            for (int v = 0; v < 8; v++) vf.u[v] = vrow[v + 8 * half];
            accO[j].v = __builtin_amdgcn_wmma_f32_16x16x32_bf16(
                false, pf.v, false, vf.v, (short)0, accO[j].v, false, false);
        }
        __syncthreads();
    }

    // epilogue: O / l, write bf16 [B,T,C] so the final GEMM reads contiguously
#pragma unroll
    for (int r = 0; r < 8; r++) {
        const float inv = 1.f / l_i[r];
        const int tq = q0 + r + 8 * half;
#pragma unroll
        for (int j = 0; j < 4; j++) {
            const int c = head * HDIM + j * 16 + col;
            attn[(size_t)(b * SEQ + tq) * EMBED + c] = cvt_bf16(accO[j].f[r] * inv);
        }
    }
}

// ---------------------------------------------------------------------------
// Kernel 3: output projection. bf16 A (attn ws) and bf16 B (pre-converted
// out_w), both async-staged; f32 output + bias.
// ---------------------------------------------------------------------------
__global__ __launch_bounds__(256)
void out_gemm(const unsigned short* __restrict__ a,
              const unsigned short* __restrict__ wb,
              const float* __restrict__ bias, float* __restrict__ out)
{
    __shared__ __align__(16) unsigned lA[128][20];
    __shared__ __align__(16) unsigned lB[64][20];
    const int tid = threadIdx.x, lane = tid & 31, wave = tid >> 5;
    const int half = lane >> 4, col = lane & 15;
    const int m0 = blockIdx.y * 128, n0 = blockIdx.x * 64;

    FragF acc[4];
    const f32x8 fzero = {};
#pragma unroll
    for (int t = 0; t < 4; t++) acc[t].v = fzero;

    for (int k0 = 0; k0 < EMBED; k0 += 32) {
#pragma unroll
        for (int i = 0; i < 2; i++) {
            int c = tid + i * 256;
            int row = c >> 2, q = c & 3;
            async_copy_b128(a + (size_t)(m0 + row) * EMBED + k0 + q * 8, &lA[row][q * 4]);
        }
        {
            int row = tid >> 2, q = tid & 3;
            async_copy_b128(wb + (size_t)(n0 + row) * EMBED + k0 + q * 8, &lB[row][q * 4]);
        }
        async_wait0();
        __syncthreads();

        FragB afr;
        const int mr = wave * 16 + col;
#pragma unroll
        for (int j = 0; j < 8; j++) {
            afr.u[j]     = lA[mr][j + 8 * half];
        }
#pragma unroll
        for (int t = 0; t < 4; t++) {
            FragB bfr; const int nr = t * 16 + col;
#pragma unroll
            for (int j = 0; j < 8; j++)
                bfr.u[j] = lB[nr][j + 8 * half];
            acc[t].v = __builtin_amdgcn_wmma_f32_16x16x32_bf16(
                false, afr.v, false, bfr.v, (short)0, acc[t].v, false, false);
        }
        __syncthreads();
    }

#pragma unroll
    for (int t = 0; t < 4; t++) {
        const int gn = n0 + t * 16 + col;
        const float bv = bias[gn];
#pragma unroll
        for (int r = 0; r < 8; r++) {
            const int gm = m0 + wave * 16 + r + 8 * half;
            out[(size_t)gm * EMBED + gn] = acc[t].f[r] + bv;
        }
    }
}

// ---------------------------------------------------------------------------
extern "C" void kernel_launch(void* const* d_in, const int* in_sizes, int n_in,
                              void* d_out, int out_size, void* d_ws, size_t ws_size,
                              hipStream_t stream)
{
    const float* x     = (const float*)d_in[0];
    const int*   mask  = (const int*)d_in[1];
    const float* qkv_w = (const float*)d_in[2];
    const float* qkv_b = (const float*)d_in[3];
    const float* out_w = (const float*)d_in[4];
    const float* out_b = (const float*)d_in[5];
    float* out = (float*)d_out;

    // workspace layout (bf16 elements)
    unsigned short* ws    = (unsigned short*)d_ws;
    unsigned short* qkv   = ws;                                  // 3*8M
    unsigned short* attn  = qkv + 3 * (size_t)HEAD_ELEMS;        // 8M
    unsigned short* xb    = attn + (size_t)HEAD_ELEMS;           // 8M
    unsigned short* wqkvb = xb + (size_t)HEAD_ELEMS;             // 3M
    unsigned short* woutb = wqkvb + 3 * (size_t)(EMBED * EMBED); // 1M

    const int n4_x = (NBATCH * SEQ * EMBED) / 4;       // 2M float4
    const int n4_wq = (3 * EMBED * EMBED) / 4;         // 768K float4
    const int n4_wo = (EMBED * EMBED) / 4;             // 256K float4

    // 0) one-shot f32 -> bf16 conversions
    f32_to_bf16<<<dim3((n4_x + 255) / 256), 256, 0, stream>>>(x, xb, n4_x);
    f32_to_bf16<<<dim3((n4_wq + 255) / 256), 256, 0, stream>>>(qkv_w, wqkvb, n4_wq);
    f32_to_bf16<<<dim3((n4_wo + 255) / 256), 256, 0, stream>>>(out_w, woutb, n4_wo);

    // 1) QKV projection: M=8192, N=3072, K=1024
    qkv_gemm<<<dim3(48, 64), 256, 0, stream>>>(xb, wqkvb, qkv_b, qkv);
    // 2) Flash attention: B*H*(T/128) blocks
    attn_kernel<<<dim3(NBATCH * HEADS * (SEQ / 128)), 256, 0, stream>>>(qkv, mask, attn);
    // 3) Output projection: M=8192, N=1024, K=1024
    out_gemm<<<dim3(16, 64), 256, 0, stream>>>(attn, woutb, out_b, out);
}
